// GPT2_42537356099871
// MI455X (gfx1250) — matrix operand order repaired and checked by reading, not compiled, not run on hardware
//
#include <hip/hip_runtime.h>
#include <hip/hip_bf16.h>

// ---------------------------------------------------------------------------
// GPT-2 small forward on MI455X (gfx1250), wave32 + WMMA bf16.
//   B=2, T=1024, D=768, H=12, HD=64, L=12, V=50257
// - All GEMMs: v_wmma_f32_16x16x32_bf16, bf16 in / f32 accumulate.
// - Global->LDS staging: GLOBAL_LOAD_ASYNC_TO_LDS_B128, double-buffered,
//   synced with s_wait_asynccnt + workgroup barrier (CDNA5 async path).
// - All B operands pre-transposed to [N][K] bf16 so staging is pure
//   contiguous 16B async lines (weights via tiled transpose-convert, V via
//   a small transpose kernel).
// - LayerNorm / softmax / residual stream stay fp32.
// ---------------------------------------------------------------------------

#define GPT_L 12
#define GPT_B 2
#define GPT_T 1024
#define GPT_D 768
#define GPT_H 12
#define GPT_HD 64
#define GPT_V 50257

typedef __attribute__((ext_vector_type(16))) __bf16 bf16x16;
typedef __attribute__((ext_vector_type(8)))  __bf16 bf16x8;
typedef __attribute__((ext_vector_type(8)))  float  f32x8;

#define BM 128
#define BN 128
#define BK 32

// flags: bit1 = GELU epilogue, bit2 = bf16 output
#define GF_GELU  2
#define GF_OBF16 4

// Per-lane async copy of 16 bytes global -> LDS (ASYNCcnt-tracked, no VGPR data).
// LDS byte offset = low 32 bits of the generic shared address (ISA aperture rule).
static __device__ __forceinline__ void async_b128(const __bf16* g, void* lds)
{
    unsigned l = (unsigned)(unsigned long long)lds;
    asm volatile("global_load_async_to_lds_b128 %0, %1, off"
                 :: "v"(l), "v"(g) : "memory");
}
static __device__ __forceinline__ void wait_async0()
{
    asm volatile("s_wait_asynccnt 0x0" ::: "memory");
}

__global__ __launch_bounds__(256) void gemm_wmma_bf16(
    const __bf16* __restrict__ A, int lda, long long saB, long long saH,
    const __bf16* __restrict__ Bm, int ldb, long long sbB, long long sbH, // B stored [N][K]
    void* __restrict__ C, int ldc, long long scB, long long scH,
    const float* __restrict__ bias, const float* __restrict__ resid,
    int M, int N, int K, int Hdim, float alpha, int flags)
{
    __shared__ __attribute__((aligned(32))) __bf16 sA[2][BM * BK];
    __shared__ __attribute__((aligned(32))) __bf16 sB[2][BN * BK]; // [n][k]

    const int bz = blockIdx.z;
    const int bb = bz / Hdim;
    const int hh = bz - bb * Hdim;
    A  += (long long)bb * saB + (long long)hh * saH;
    Bm += (long long)bb * sbB + (long long)hh * sbH;
    const long long coff = (long long)bb * scB + (long long)hh * scH;
    float*  Cf = (float*)C  + coff;
    __bf16* Cb = (__bf16*)C + coff;
    const float* Rf = resid ? (resid + coff) : nullptr;

    const int m0  = blockIdx.x * BM;
    const int n0  = blockIdx.y * BN;
    const int tid = threadIdx.x;
    const int lane  = tid & 31;
    const int wave  = tid >> 5;
    const int wm    = (wave >> 2) * 64;  // 0 or 64
    const int wn    = (wave & 3) * 32;   // 0..96
    const int lhalf = (lane < 16) ? 0 : 1;
    const int lmod  = lane & 15;

    // Issue async staging of one K-tile into buffer `buf`.
    // Each thread owns 2 chunks of 8 bf16 (16B) for A and the same for B.
    auto stage = [&](int k0, int buf) {
#pragma unroll
        for (int c = tid; c < (BM * BK) / 8; c += 256) {
            int row = c >> 2;
            int kc  = (c & 3) << 3;
            if (m0 + row < M)
                async_b128(A + (long long)(m0 + row) * lda + k0 + kc,
                           &sA[buf][row * BK + kc]);
            if (n0 + row < N)
                async_b128(Bm + (long long)(n0 + row) * ldb + k0 + kc,
                           &sB[buf][row * BK + kc]);
            // OOB tiles skipped: stale LDS only feeds rows/cols the epilogue discards.
        }
    };

    f32x8 acc[4][2] = {};

    stage(0, 0);
    wait_async0();
    __syncthreads();

    int buf = 0;
    for (int k0 = 0; k0 < K; k0 += BK) {
        if (k0 + BK < K) stage(k0 + BK, buf ^ 1);   // prefetch next tile (async)

        // ---- fragments in contiguous k order, same for A and B ----
        const int kbA = lhalf ? 16 : 0;  // A: lanes 0-15 -> K 0..15 ; 16-31 -> K 16..31 (contiguous)
        const int kbB = lhalf ? 16 : 0;  // B: lanes 0-15 -> K 0..15 ; 16-31 -> K 16..31 (contiguous)

        bf16x16 bfr[2];
#pragma unroll
        for (int nt = 0; nt < 2; nt++)
            bfr[nt] = *(const bf16x16*)(&sB[buf][(wn + nt * 16 + lmod) * BK + kbB]);

#pragma unroll
        for (int mt = 0; mt < 4; mt++) {
            int row = wm + mt * 16 + lmod;
            bf16x8 lo = *(const bf16x8*)(&sA[buf][row * BK + kbA]);
            bf16x8 hi = *(const bf16x8*)(&sA[buf][row * BK + kbA + 8]);
            bf16x16 afr;
#pragma unroll
            for (int i = 0; i < 8; i++) { afr[i] = lo[i]; afr[8 + i] = hi[i]; }
#pragma unroll
            for (int nt = 0; nt < 2; nt++) {
                acc[mt][nt] = __builtin_amdgcn_wmma_f32_16x16x32_bf16(
                    false, afr, false, bfr[nt], (short)0, acc[mt][nt], false, false);
            }
        }

        wait_async0();      // next buffer's lines have landed
        __syncthreads();    // all waves done reading current buffer + all loads visible
        buf ^= 1;
    }

    // ---- epilogue: C layout = VGPR r -> row (r or r+8 by lane half), col = lane&15 ----
#pragma unroll
    for (int mt = 0; mt < 4; mt++) {
#pragma unroll
        for (int nt = 0; nt < 2; nt++) {
            int col = n0 + wn + nt * 16 + lmod;
            if (col >= N) continue;
            float bv = bias ? bias[col] : 0.0f;
#pragma unroll
            for (int r = 0; r < 8; r++) {
                int row = m0 + wm + mt * 16 + r + lhalf * 8;
                if (row >= M) continue;
                float v = acc[mt][nt][r] * alpha + bv;
                if (Rf) v += Rf[(long long)row * ldc + col];
                if (flags & GF_GELU) {
                    float u = 0.7978845608028654f * (v + 0.044715f * v * v * v);
                    v = 0.5f * v * (1.0f + tanhf(u));
                }
                if (flags & GF_OBF16) Cb[(long long)row * ldc + col] = (__bf16)v;
                else                  Cf[(long long)row * ldc + col] = v;
            }
        }
    }
}

// ---------------------------------------------------------------------------
__global__ __launch_bounds__(256) void embed_kernel(
    const int* __restrict__ idx, const float* __restrict__ wte,
    const float* __restrict__ wpe, float* __restrict__ x)
{
    int row = blockIdx.x;                 // 0..B*T-1
    int t   = row % GPT_T;
    int tok = idx[row];
    const float* we = wte + (long long)tok * GPT_D;
    const float* wp = wpe + (long long)t * GPT_D;
    float* o = x + (long long)row * GPT_D;
    for (int j = threadIdx.x; j < GPT_D; j += 256) o[j] = we[j] + wp[j];
}

__global__ __launch_bounds__(256) void layernorm_bf16_kernel(
    const float* __restrict__ x, const float* __restrict__ g,
    const float* __restrict__ b, __bf16* __restrict__ out)
{
    __shared__ float red[256];
    const int row = blockIdx.x;
    const int tid = threadIdx.x;
    const float* xr = x + (long long)row * GPT_D;
    float v[3];
    float s = 0.0f;
#pragma unroll
    for (int c = 0; c < 3; c++) { v[c] = xr[tid + c * 256]; s += v[c]; }
    red[tid] = s; __syncthreads();
    for (int st = 128; st > 0; st >>= 1) { if (tid < st) red[tid] += red[tid + st]; __syncthreads(); }
    float mu = red[0] * (1.0f / GPT_D); __syncthreads();
    s = 0.0f;
#pragma unroll
    for (int c = 0; c < 3; c++) { float d = v[c] - mu; s += d * d; }
    red[tid] = s; __syncthreads();
    for (int st = 128; st > 0; st >>= 1) { if (tid < st) red[tid] += red[tid + st]; __syncthreads(); }
    float rstd = rsqrtf(red[0] * (1.0f / GPT_D) + 1e-5f);
    __bf16* o = out + (long long)row * GPT_D;
#pragma unroll
    for (int c = 0; c < 3; c++) {
        int j = tid + c * 256;
        o[j] = (__bf16)((v[c] - mu) * rstd * g[j] + b[j]);
    }
}

// In-place causal softmax over bf16 score rows; one block per (b,h,i) row.
__global__ __launch_bounds__(256) void softmax_causal_kernel(__bf16* __restrict__ P)
{
    __shared__ float red[256];
    const int bid = blockIdx.x;
    const int i   = bid % GPT_T;
    const long long bh = bid / GPT_T;
    __bf16* row = P + bh * (long long)GPT_T * GPT_T + (long long)i * GPT_T;
    const int tid = threadIdx.x;
    float vals[4];
    float mx = -3.4e38f;
#pragma unroll
    for (int c = 0; c < 4; c++) {
        int j = tid + c * 256;
        float v = (j <= i) ? (float)row[j] : -3.4e38f;
        vals[c] = v; mx = fmaxf(mx, v);
    }
    red[tid] = mx; __syncthreads();
    for (int st = 128; st > 0; st >>= 1) { if (tid < st) red[tid] = fmaxf(red[tid], red[tid + st]); __syncthreads(); }
    mx = red[0]; __syncthreads();
    float sum = 0.0f;
#pragma unroll
    for (int c = 0; c < 4; c++) {
        int j = tid + c * 256;
        float e = (j <= i) ? __expf(vals[c] - mx) : 0.0f;
        vals[c] = e; sum += e;
    }
    red[tid] = sum; __syncthreads();
    for (int st = 128; st > 0; st >>= 1) { if (tid < st) red[tid] += red[tid + st]; __syncthreads(); }
    float inv = 1.0f / red[0];
#pragma unroll
    for (int c = 0; c < 4; c++) {
        int j = tid + c * 256;
        row[j] = (__bf16)(vals[c] * inv);   // j>i written as exact 0
    }
}

// Tiled transpose + fp32->bf16 convert: in [K][N] fp32 -> out [N][K] bf16.
__global__ __launch_bounds__(256) void transpose_convert_kernel(
    const float* __restrict__ in, __bf16* __restrict__ out, int K, int N)
{
    __shared__ float tile[32][33];
    const int kt = blockIdx.x * 32;
    const int nt = blockIdx.y * 32;
    const int tx = threadIdx.x & 31;
    const int ty = threadIdx.x >> 5;   // 0..7
#pragma unroll
    for (int r = 0; r < 32; r += 8) {
        int k = kt + ty + r, n = nt + tx;
        tile[ty + r][tx] = (k < K && n < N) ? in[(long long)k * N + n] : 0.0f;
    }
    __syncthreads();
#pragma unroll
    for (int r = 0; r < 32; r += 8) {
        int n = nt + ty + r, k = kt + tx;
        if (n < N && k < K) out[(long long)n * K + k] = (__bf16)tile[tx][ty + r];
    }
}

// vT[(b*H+h)][d][t] = qkv[b][t][2D + h*HD + d]   (bf16 -> bf16)
__global__ __launch_bounds__(256) void vtrans_kernel(
    const __bf16* __restrict__ qkv, __bf16* __restrict__ vT)
{
    long long i = (long long)blockIdx.x * 256 + threadIdx.x;  // over B*H*HD*T
    int t = (int)(i & (GPT_T - 1));
    long long r = i >> 10;
    int d  = (int)(r % GPT_HD);
    long long bh = r / GPT_HD;
    int h = (int)(bh % GPT_H);
    int b = (int)(bh / GPT_H);
    vT[i] = qkv[((long long)b * GPT_T + t) * (3 * GPT_D) + 2 * GPT_D + h * GPT_HD + d];
}

// ---------------------------------------------------------------------------
static inline void launch_tconv(hipStream_t s, const float* in, __bf16* out, int K, int N) {
    dim3 g((K + 31) / 32, (N + 31) / 32, 1);
    transpose_convert_kernel<<<g, 256, 0, s>>>(in, out, K, N);
}

static inline void launch_gemm(hipStream_t s,
    const __bf16* A, int lda, long long saB, long long saH,
    const __bf16* B, int ldb, long long sbB, long long sbH,
    void* C, int ldc, long long scB, long long scH,
    const float* bias, const float* resid,
    int M, int N, int K, int Hdim, int Z, float alpha, int flags)
{
    dim3 g((M + BM - 1) / BM, (N + BN - 1) / BN, Z);
    gemm_wmma_bf16<<<g, dim3(256), 0, s>>>(A, lda, saB, saH, B, ldb, sbB, sbH,
                                           C, ldc, scB, scH, bias, resid,
                                           M, N, K, Hdim, alpha, flags);
}

extern "C" void kernel_launch(void* const* d_in, const int* in_sizes, int n_in,
                              void* d_out, int out_size, void* d_ws, size_t ws_size,
                              hipStream_t stream)
{
    (void)in_sizes; (void)n_in; (void)out_size; (void)ws_size;

    const int*   idx     = (const int*)  d_in[0];
    const float* wte     = (const float*)d_in[1];
    const float* wpe     = (const float*)d_in[2];
    const float* ln1_g   = (const float*)d_in[3];
    const float* ln1_b   = (const float*)d_in[4];
    const float* attn_w  = (const float*)d_in[5];
    const float* attn_b  = (const float*)d_in[6];
    const float* aproj_w = (const float*)d_in[7];
    const float* aproj_b = (const float*)d_in[8];
    const float* ln2_g   = (const float*)d_in[9];
    const float* ln2_b   = (const float*)d_in[10];
    const float* fc_w    = (const float*)d_in[11];
    const float* fc_b    = (const float*)d_in[12];
    const float* mproj_w = (const float*)d_in[13];
    const float* mproj_b = (const float*)d_in[14];
    const float* lnf_g   = (const float*)d_in[15];
    const float* lnf_b   = (const float*)d_in[16];
    const float* lm_w    = (const float*)d_in[17];

    const int BT_ = GPT_B * GPT_T;           // 2048 rows
    const int D3  = 3 * GPT_D;               // 2304
    const int D4  = 4 * GPT_D;               // 3072

    // ---- workspace carve (256B aligned) ----
    char* base = (char*)d_ws;
    size_t off = 0;
    auto carve = [&](size_t bytes) -> void* {
        void* p = base + off;
        off = (off + bytes + 255) & ~(size_t)255;
        return p;
    };
    float*  xa   = (float*) carve((size_t)BT_ * GPT_D * 4);
    float*  xb   = (float*) carve((size_t)BT_ * GPT_D * 4);
    __bf16* hb   = (__bf16*)carve((size_t)BT_ * GPT_D * 2);
    __bf16* qkv  = (__bf16*)carve((size_t)BT_ * D3 * 2);
    __bf16* Pm   = (__bf16*)carve((size_t)GPT_B * GPT_H * GPT_T * GPT_T * 2);
    __bf16* obuf = (__bf16*)carve((size_t)BT_ * GPT_D * 2);
    __bf16* fbuf = (__bf16*)carve((size_t)BT_ * D4 * 2);
    __bf16* vT   = (__bf16*)carve((size_t)GPT_B * GPT_H * GPT_HD * GPT_T * 2);
    __bf16* wbuf = (__bf16*)carve((size_t)GPT_D * GPT_V * 2); // per-layer pack / lm head (transposed)

    // per-layer transposed bf16 weight slots inside wbuf ([N][K] layouts)
    __bf16* w_attnT  = wbuf;                                    // [3D][D]
    __bf16* w_aprojT = w_attnT  + (size_t)GPT_D * D3;           // [D][D]
    __bf16* w_fcT    = w_aprojT + (size_t)GPT_D * GPT_D;        // [4D][D]
    __bf16* w_mprojT = w_fcT    + (size_t)GPT_D * D4;           // [D][4D]

    // ---- embedding ----
    embed_kernel<<<BT_, 256, 0, stream>>>(idx, wte, wpe, xa);

    const long long sQb = (long long)GPT_T * D3;   // per-batch row stride in qkv
    const long long sPb = (long long)GPT_H * GPT_T * GPT_T;
    const long long sPh = (long long)GPT_T * GPT_T;
    const long long sVb = (long long)GPT_H * GPT_HD * GPT_T;
    const long long sVh = (long long)GPT_HD * GPT_T;

    for (int l = 0; l < GPT_L; l++) {
        // convert this layer's weights fp32 [K][N] -> bf16 [N][K]
        launch_tconv(stream, attn_w  + (long long)l * GPT_D * D3,    w_attnT,  GPT_D, D3);
        launch_tconv(stream, aproj_w + (long long)l * GPT_D * GPT_D, w_aprojT, GPT_D, GPT_D);
        launch_tconv(stream, fc_w    + (long long)l * GPT_D * D4,    w_fcT,    GPT_D, D4);
        launch_tconv(stream, mproj_w + (long long)l * D4 * GPT_D,    w_mprojT, D4, GPT_D);

        // h = LN1(x)
        layernorm_bf16_kernel<<<BT_, 256, 0, stream>>>(xa, ln1_g + l * GPT_D, ln1_b + l * GPT_D, hb);

        // qkv = h @ attn_w + attn_b  -> bf16 [BT, 3D]
        launch_gemm(stream, hb, GPT_D, 0, 0, w_attnT, GPT_D, 0, 0, qkv, D3, 0, 0,
                    attn_b + (long long)l * D3, nullptr,
                    BT_, D3, GPT_D, 1, 1, 1.0f, GF_OBF16);

        // vT = transpose of V heads
        vtrans_kernel<<<(GPT_B * GPT_H * GPT_HD * GPT_T) / 256, 256, 0, stream>>>(qkv, vT);

        // scores = alpha * q @ k^T  (per b,h; k rows are already [N][K])
        launch_gemm(stream,
                    qkv,         D3, sQb, GPT_HD,          // A = q
                    qkv + GPT_D, D3, sQb, GPT_HD,          // B = k, [T][HD]
                    Pm, GPT_T, sPb, sPh,
                    nullptr, nullptr,
                    GPT_T, GPT_T, GPT_HD, GPT_H, GPT_B * GPT_H,
                    0.125f /* 1/sqrt(64) */, GF_OBF16);

        // causal softmax in-place on Pm
        softmax_causal_kernel<<<GPT_B * GPT_H * GPT_T, 256, 0, stream>>>(Pm);

        // O = P @ V  -> bf16 [B,T,D]   (B operand = vT [HD][T])
        launch_gemm(stream,
                    Pm, GPT_T, sPb, sPh,
                    vT, GPT_T, sVb, sVh,
                    obuf, GPT_D, (long long)GPT_T * GPT_D, GPT_HD,
                    nullptr, nullptr,
                    GPT_T, GPT_HD, GPT_T, GPT_H, GPT_B * GPT_H,
                    1.0f, GF_OBF16);

        // x = x + O @ aproj_w + aproj_b  -> fp32 xb
        launch_gemm(stream, obuf, GPT_D, 0, 0, w_aprojT, GPT_D, 0, 0, xb, GPT_D, 0, 0,
                    aproj_b + (long long)l * GPT_D, xa,
                    BT_, GPT_D, GPT_D, 1, 1, 1.0f, 0);

        // h = LN2(x)
        layernorm_bf16_kernel<<<BT_, 256, 0, stream>>>(xb, ln2_g + l * GPT_D, ln2_b + l * GPT_D, hb);

        // f = gelu(h @ fc_w + fc_b) -> bf16
        launch_gemm(stream, hb, GPT_D, 0, 0, w_fcT, GPT_D, 0, 0, fbuf, D4, 0, 0,
                    fc_b + (long long)l * D4, nullptr,
                    BT_, D4, GPT_D, 1, 1, 1.0f, GF_GELU | GF_OBF16);

        // x = x + f @ mproj_w + mproj_b -> fp32 xa
        launch_gemm(stream, fbuf, D4, 0, 0, w_mprojT, D4, 0, 0, xa, GPT_D, 0, 0,
                    mproj_b + (long long)l * GPT_D, xb,
                    BT_, GPT_D, D4, 1, 1, 1.0f, 0);
    }

    // final LN + lm_head (lm_w [D][V] -> transposed [V][D] bf16)
    launch_tconv(stream, lm_w, wbuf, GPT_D, GPT_V);
    layernorm_bf16_kernel<<<BT_, 256, 0, stream>>>(xa, lnf_g, lnf_b, hb);
    launch_gemm(stream, hb, GPT_D, 0, 0, wbuf, GPT_D, 0, 0, (float*)d_out, GPT_V, 0, 0,
                nullptr, nullptr,
                BT_, GPT_V, GPT_D, 1, 1, 1.0f, 0);
}
